// HarmonicSlotEncoder_62423054680415
// MI455X (gfx1250) — compile-verified, matrix operand order checked by reading, not run on hardware
//
#include <hip/hip_runtime.h>
#include <hip/hip_bf16.h>
#include <math.h>

// ---- problem constants -------------------------------------------------
#define Bq   32      // batch
#define Cq   3       // channels
#define IMGq 512
#define Pq   16      // patch
#define Dq   256     // model dim
#define Sq   8       // slots
#define Nq   1024    // patches per image (32x32)
#define Kpatch 768   // C*P*P

typedef __attribute__((ext_vector_type(16))) __bf16 v16bf;
typedef __attribute__((ext_vector_type(8)))  __bf16 v8bf;
typedef __attribute__((ext_vector_type(8)))  float  v8f;
typedef __attribute__((ext_vector_type(4)))  float  v4f;

__device__ __forceinline__ float gelu_erf(float x) {
    return 0.5f * x * (1.0f + erff(x * 0.70710678118654752440f));
}

// Build a v16bf fragment from four float4 runs (pure convert, no adds).
__device__ __forceinline__ v16bf cvt16(v4f a, v4f b, v4f c, v4f d) {
    v16bf r;
#pragma unroll
    for (int i = 0; i < 4; ++i) {
        r[i]      = (__bf16)a[i];
        r[4 + i]  = (__bf16)b[i];
        r[8 + i]  = (__bf16)c[i];
        r[12 + i] = (__bf16)d[i];
    }
    return r;
}

// Same but with a real epsilon added before conversion (upd kernel only).
__device__ __forceinline__ v16bf cvt16_eps(v4f a, v4f b, v4f c, v4f d, float eps) {
    v16bf r;
#pragma unroll
    for (int i = 0; i < 4; ++i) {
        r[i]      = (__bf16)(a[i] + eps);
        r[4 + i]  = (__bf16)(b[i] + eps);
        r[8 + i]  = (__bf16)(c[i] + eps);
        r[12 + i] = (__bf16)(d[i] + eps);
    }
    return r;
}

// A-fragment (fp32, row-major, k contiguous). base = row_ptr + kb + 8*half.
// ISA 7.12.2: lane holds k runs [kb+8h, +8) and [kb+16+8h, +8).
__device__ __forceinline__ v16bf load_a_frag(const float* __restrict__ base) {
    const v4f* p0 = (const v4f*)base;
    const v4f* p1 = (const v4f*)(base + 16);
    return cvt16(p0[0], p0[1], p1[0], p1[1]);
}

// B-fragment (bf16 packed [N,K], k contiguous). base = col_ptr + kb + 8*half.
__device__ __forceinline__ v16bf load_b_frag(const __bf16* __restrict__ base) {
    v8bf lo = *(const v8bf*)base;
    v8bf hi = *(const v8bf*)(base + 16);
    return __builtin_shufflevector(lo, hi, 0,1,2,3,4,5,6,7,8,9,10,11,12,13,14,15);
}

// ---- weight pre-pack: fp32 [K,N] or [N,K] -> bf16 [N,K] ----------------
__global__ void pack_w_kernel(const float* __restrict__ W, __bf16* __restrict__ Wp,
                              int K, int N, int srcNK)
{
    int idx = blockIdx.x * blockDim.x + threadIdx.x;
    if (idx >= N * K) return;
    int n = idx / K, k = idx % K;
    Wp[idx] = (__bf16)(srcNK ? W[idx] : W[(size_t)k * N + n]);
}

// ---- generic WMMA GEMM: out[M,N] = A[M,K] @ Wp^T (+bias,+addSrc,gelu) --
// Wp is bf16 packed [N,K]. One wave per 16x32 tile (A fragment reused x2).
__global__ void wmma_gemm_kernel(const float* __restrict__ A,
                                 const __bf16* __restrict__ Wp,
                                 const float* __restrict__ bias,
                                 const float* __restrict__ addSrc,
                                 float* __restrict__ out,
                                 int M, int Ncols, int K, int doGelu)
{
    const int lane = threadIdx.x & 31;
    const int half = lane >> 4;
    const int lm   = lane & 15;
    const int row  = blockIdx.y * 16 + lm;
    const int col0 = blockIdx.x * 32 + lm;
    const float*  __restrict__ arow  = A  + (size_t)row  * K + 8 * half;
    const __bf16* __restrict__ wrow0 = Wp + (size_t)col0 * K + 8 * half;
    const __bf16* __restrict__ wrow1 = wrow0 + (size_t)16 * K;

    v8f acc0 = {}, acc1 = {};
#pragma unroll 2
    for (int kb = 0; kb < K; kb += 32) {
        __builtin_prefetch(arow + kb + 256, 0, 1);  // global_prefetch_b8
        v16bf af  = load_a_frag(arow + kb);
        v16bf bf0 = load_b_frag(wrow0 + kb);
        v16bf bf1 = load_b_frag(wrow1 + kb);
        acc0 = __builtin_amdgcn_wmma_f32_16x16x32_bf16(
                   false, af, false, bf0, (short)0, acc0, false, false);
        acc1 = __builtin_amdgcn_wmma_f32_16x16x32_bf16(
                   false, af, false, bf1, (short)0, acc1, false, false);
    }
#pragma unroll
    for (int r = 0; r < 8; ++r) {
        int m = blockIdx.y * 16 + r + 8 * half;
#pragma unroll
        for (int t = 0; t < 2; ++t) {
            int col = col0 + 16 * t;
            size_t idx = (size_t)m * Ncols + col;
            float v = t ? acc1[r] : acc0[r];
            if (bias)   v += bias[col];
            if (addSrc) v += addSrc[idx];
            if (doGelu) v = gelu_erf(v);
            out[idx] = v;
        }
    }
}

// ---- patch embed: im2col GEMM  x0[BN,D] = patches @ conv_w^T + b + pos -
// Each 8-long k-run stays inside one pixel row of a patch -> float4 loads.
// 16x32 tile per wave: the im2col A gather is reused for 2 output tiles.
__global__ void patch_embed_kernel(const float* __restrict__ state,
                                   const __bf16* __restrict__ convp, // bf16 [D, 768]
                                   const float* __restrict__ convb,
                                   const float* __restrict__ pos,    // [N, D]
                                   float* __restrict__ x0)
{
    const int lane = threadIdx.x & 31;
    const int half = lane >> 4;
    const int lm   = lane & 15;
    const int row  = blockIdx.y * 16 + lm;        // b*N + p
    const int b    = row >> 10;
    const int p    = row & 1023;
    const int py   = p >> 5, px = p & 31;
    const int col0 = blockIdx.x * 32 + lm;        // output channel (first tile)
    const float*  __restrict__ sb    = state + (size_t)b * Cq * IMGq * IMGq
                                             + (size_t)(py * Pq) * IMGq + px * Pq;
    const __bf16* __restrict__ wrow0 = convp + (size_t)col0 * Kpatch + 8 * half;
    const __bf16* __restrict__ wrow1 = wrow0 + (size_t)16 * Kpatch;

    v8f acc0 = {}, acc1 = {};
    for (int kb = 0; kb < Kpatch; kb += 32) {
        int k0 = kb + 8 * half;       // first run start
        int k1 = k0 + 16;             // second run start
        int c0 = k0 >> 8, r0 = (k0 >> 4) & 15, x0i = k0 & 15;
        int c1 = k1 >> 8, r1 = (k1 >> 4) & 15, x1i = k1 & 15;
        const v4f* p0 = (const v4f*)(sb + ((size_t)c0 * IMGq + r0) * IMGq + x0i);
        const v4f* p1 = (const v4f*)(sb + ((size_t)c1 * IMGq + r1) * IMGq + x1i);
        v16bf af  = cvt16(p0[0], p0[1], p1[0], p1[1]);
        v16bf bf0 = load_b_frag(wrow0 + kb);
        v16bf bf1 = load_b_frag(wrow1 + kb);
        acc0 = __builtin_amdgcn_wmma_f32_16x16x32_bf16(
                   false, af, false, bf0, (short)0, acc0, false, false);
        acc1 = __builtin_amdgcn_wmma_f32_16x16x32_bf16(
                   false, af, false, bf1, (short)0, acc1, false, false);
    }
#pragma unroll
    for (int r = 0; r < 8; ++r) {
        int m  = blockIdx.y * 16 + r + 8 * half;
        int pp = m & 1023;
#pragma unroll
        for (int t = 0; t < 2; ++t) {
            int col = col0 + 16 * t;
            float v = t ? acc1[r] : acc0[r];
            x0[(size_t)m * Dq + col] = v + convb[col] + pos[(size_t)pp * Dq + col];
        }
    }
}

// ---- LayerNorm (one wave per row), optional erf-GELU -------------------
__global__ void ln_kernel(const float* __restrict__ in, float* __restrict__ out,
                          const float* __restrict__ g, const float* __restrict__ b,
                          int Nd, int doGelu)
{
    const int row  = blockIdx.x;
    const int lane = threadIdx.x;
    const float* __restrict__ x = in + (size_t)row * Nd;
    float s = 0.f;
    for (int i = lane; i < Nd; i += 32) s += x[i];
#pragma unroll
    for (int m = 16; m; m >>= 1) s += __shfl_xor(s, m, 32);
    float mean = s / (float)Nd;
    float v = 0.f;
    for (int i = lane; i < Nd; i += 32) { float d = x[i] - mean; v += d * d; }
#pragma unroll
    for (int m = 16; m; m >>= 1) v += __shfl_xor(v, m, 32);
    float rstd = rsqrtf(v / (float)Nd + 1e-5f);
    float* __restrict__ o = out + (size_t)row * Nd;
    for (int i = lane; i < Nd; i += 32) {
        float y = (x[i] - mean) * rstd * g[i] + b[i];
        if (doGelu) y = gelu_erf(y);
        o[i] = y;
    }
}

// ---- logits[b,n,s] = (k[b,n,:] . q[b,s,:]) * D^-0.5  (S=8 padded to 16)
__global__ void logits_kernel(const float* __restrict__ kmat,  // [B,N,D]
                              const float* __restrict__ q,     // [B,S,D]
                              float* __restrict__ logits)      // [B,N,S]
{
    const int lane = threadIdx.x & 31;
    const int half = lane >> 4;
    const int lm   = lane & 15;
    const int b    = blockIdx.y;
    const float* __restrict__ arow =
        kmat + ((size_t)b * Nq + blockIdx.x * 16 + lm) * Dq + 8 * half;
    const float* __restrict__ qrow =
        q + ((size_t)b * Sq + (lm & 7)) * Dq + 8 * half;   // clamped row, guarded below

    v8f acc = {};
    for (int kb = 0; kb < Dq; kb += 32) {
        v16bf af = load_a_frag(arow + kb);
        v4f z = {0.f, 0.f, 0.f, 0.f};
        v4f q0 = z, q1 = z, q2 = z, q3 = z;
        if (lm < Sq) {
            const v4f* p0 = (const v4f*)(qrow + kb);
            const v4f* p1 = (const v4f*)(qrow + kb + 16);
            q0 = p0[0]; q1 = p0[1]; q2 = p1[0]; q3 = p1[1];
        }
        v16bf bf = cvt16(q0, q1, q2, q3);
        acc = __builtin_amdgcn_wmma_f32_16x16x32_bf16(
                  false, af, false, bf, (short)0, acc, false, false);
    }
#pragma unroll
    for (int r = 0; r < 8; ++r) {
        int n = blockIdx.x * 16 + r + 8 * half;
        if (lm < Sq)
            logits[((size_t)b * Nq + n) * Sq + lm] = acc[r] * 0.0625f;  // 256^-0.5
    }
}

// ---- softmax over slots; writes TRANSPOSED attn [B,S,N] (= masks layout)
__global__ void softmax_kernel(const float* __restrict__ logits,
                               float* __restrict__ attT)
{
    int idx = blockIdx.x * blockDim.x + threadIdx.x;     // b*N + n
    if (idx >= Bq * Nq) return;
    int b = idx >> 10, n = idx & 1023;
    const v4f* p = (const v4f*)(logits + (size_t)idx * Sq);
    v4f lo = p[0], hi = p[1];
    float v[Sq];
#pragma unroll
    for (int s = 0; s < 4; ++s) { v[s] = lo[s]; v[4 + s] = hi[s]; }
    float mx = v[0];
#pragma unroll
    for (int s = 1; s < Sq; ++s) mx = fmaxf(mx, v[s]);
    float sum = 0.f;
#pragma unroll
    for (int s = 0; s < Sq; ++s) { v[s] = expf(v[s] - mx); sum += v[s]; }
    float inv = 1.0f / sum;
#pragma unroll
    for (int s = 0; s < Sq; ++s)
        attT[((size_t)(b * Sq + s)) * Nq + n] = v[s] * inv;
}

// ---- colsum[b,s] = sum_n (attT[b,s,n] + 1e-8)  (contiguous rows) -------
__global__ void colsum_kernel(const float* __restrict__ attT, float* __restrict__ colsum)
{
    __shared__ float sm[256];
    int bs = blockIdx.x;                 // b*S + s
    const float* __restrict__ rowp = attT + (size_t)bs * Nq;
    float acc = 0.f;
    for (int n = threadIdx.x; n < Nq; n += 256) acc += rowp[n] + 1e-8f;
    sm[threadIdx.x] = acc;
    __syncthreads();
    for (int st = 128; st; st >>= 1) {
        if (threadIdx.x < st) sm[threadIdx.x] += sm[threadIdx.x + st];
        __syncthreads();
    }
    if (threadIdx.x == 0) colsum[bs] = sm[0];
}

// ---- upd[b,s,:] = (1/colsum) * sum_n (attT[b,s,n]+1e-8) * v[b,n,:] -----
// A rows contiguous (attT); B loads lane-coalesced over d.
__global__ void upd_kernel(const float* __restrict__ attT,   // [B,S,N]
                           const float* __restrict__ vmat,   // [B,N,D]
                           const float* __restrict__ colsum, // [B*S]
                           float* __restrict__ upd)          // [B,S,D]
{
    const int lane = threadIdx.x & 31;
    const int half = lane >> 4;
    const int lm   = lane & 15;
    const int b    = blockIdx.y;
    const int col  = blockIdx.x * 16 + lm;   // D column
    const float* __restrict__ arow =
        attT + ((size_t)b * Sq + (lm & 7)) * Nq + 8 * half;  // guarded below
    const float* __restrict__ vb = vmat + (size_t)b * Nq * Dq;

    v8f acc = {};
    for (int kb = 0; kb < Nq; kb += 32) {
        v4f z = {0.f, 0.f, 0.f, 0.f};
        v4f a0 = z, a1 = z, a2 = z, a3 = z;
        if (lm < Sq) {
            const v4f* p0 = (const v4f*)(arow + kb);
            const v4f* p1 = (const v4f*)(arow + kb + 16);
            a0 = p0[0]; a1 = p0[1]; a2 = p1[0]; a3 = p1[1];
        }
        v16bf af = cvt16_eps(a0, a1, a2, a3, 1e-8f);   // +eps (padded rows unused)
        v16bf bf;
#pragma unroll
        for (int e = 0; e < 16; ++e) {
            int k = kb + e + 8 * ((e >> 3) + half);
            bf[e] = (__bf16)vb[(size_t)k * Dq + col];   // lane-coalesced over col
        }
        acc = __builtin_amdgcn_wmma_f32_16x16x32_bf16(
                  false, af, false, bf, (short)0, acc, false, false);
    }
#pragma unroll
    for (int r = 0; r < 8; ++r) {
        int s = r + 8 * half;                // C row = slot index
        if (s < Sq)
            upd[((size_t)(b * Sq + s)) * Dq + col] = acc[r] / colsum[b * Sq + s];
    }
}

// ---- GRU combine (torch gate order r,z,n); slots updated in place ------
__global__ void gru_kernel(const float* __restrict__ gi, const float* __restrict__ gh,
                           float* __restrict__ slots)
{
    int idx = blockIdx.x * blockDim.x + threadIdx.x;
    if (idx >= Bq * Sq * Dq) return;
    int row = idx / Dq, d = idx % Dq;
    const float* __restrict__ gir = gi + (size_t)row * 3 * Dq;
    const float* __restrict__ ghr = gh + (size_t)row * 3 * Dq;
    float r  = 1.0f / (1.0f + expf(-(gir[d]          + ghr[d])));
    float z  = 1.0f / (1.0f + expf(-(gir[Dq + d]     + ghr[Dq + d])));
    float ng = tanhf(gir[2 * Dq + d] + r * ghr[2 * Dq + d]);
    float prev = slots[idx];
    slots[idx] = (1.0f - z) * ng + z * prev;
}

__global__ void init_slots_kernel(const float* __restrict__ priors,
                                  const float* __restrict__ mu,
                                  float* __restrict__ slots)
{
    int idx = blockIdx.x * blockDim.x + threadIdx.x;
    if (idx >= Bq * Sq * Dq) return;
    slots[idx] = priors[idx % (Sq * Dq)] + mu[idx % Dq];
}

__global__ void copy_kernel(const float* __restrict__ src, float* __restrict__ dst, int n)
{
    int i = blockIdx.x * blockDim.x + threadIdx.x;
    if (i < n) dst[i] = src[i];
}

// ------------------------------------------------------------------------
extern "C" void kernel_launch(void* const* d_in, const int* in_sizes, int n_in,
                              void* d_out, int out_size, void* d_ws, size_t ws_size,
                              hipStream_t stream)
{
    const float* state    = (const float*)d_in[0];
    const float* conv_w   = (const float*)d_in[1];
    const float* conv_b   = (const float*)d_in[2];
    const float* pos      = (const float*)d_in[3];
    const float* pmlp_w1  = (const float*)d_in[4];
    const float* pmlp_b1  = (const float*)d_in[5];
    const float* pmlp_lng = (const float*)d_in[6];
    const float* pmlp_lnb = (const float*)d_in[7];
    const float* pmlp_w2  = (const float*)d_in[8];
    const float* pmlp_b2  = (const float*)d_in[9];
    const float* priors   = (const float*)d_in[10];
    const float* slot_mu  = (const float*)d_in[11];
    const float* ln_in_g  = (const float*)d_in[13];
    const float* ln_in_b  = (const float*)d_in[14];
    const float* ln_sl_g  = (const float*)d_in[15];
    const float* ln_sl_b  = (const float*)d_in[16];
    const float* ln_mlp_g = (const float*)d_in[17];
    const float* ln_mlp_b = (const float*)d_in[18];
    const float* q_w      = (const float*)d_in[19];
    const float* k_w      = (const float*)d_in[20];
    const float* v_w      = (const float*)d_in[21];
    const float* gru_wih  = (const float*)d_in[22];
    const float* gru_whh  = (const float*)d_in[23];
    const float* gru_bih  = (const float*)d_in[24];
    const float* gru_bhh  = (const float*)d_in[25];
    const float* smlp_w1  = (const float*)d_in[26];
    const float* smlp_b1  = (const float*)d_in[27];
    const float* smlp_w2  = (const float*)d_in[28];
    const float* smlp_b2  = (const float*)d_in[29];

    float* out_slots = (float*)d_out;                 // [B,S,D] = 65536
    float* attT      = (float*)d_out + Bq * Sq * Dq;  // [B,S,N] = masks layout

    const int BN = Bq * Nq;       // 32768 token rows
    const int BS = Bq * Sq;       // 256 slot rows

    // workspace layout (floats, all offsets 16B-aligned)
    float* ws    = (float*)d_ws;
    float* x0    = ws;                                 // [BN, D]   (also final x)
    float* h     = x0    + (size_t)BN * Dq;            // [BN, 2D]
    float* kbuf  = h     + (size_t)BN * 2 * Dq;        // [BN, D]
    float* vbuf  = kbuf  + (size_t)BN * Dq;            // [BN, D]
    float* slots = vbuf  + (size_t)BN * Dq;            // [BS, D]
    float* snorm = slots + (size_t)BS * Dq;            // [BS, D]
    float* qbuf  = snorm + (size_t)BS * Dq;            // [BS, D]
    float* logit = qbuf  + (size_t)BS * Dq;            // [B, N, S]
    float* csum  = logit + (size_t)Bq * Nq * Sq;       // [BS]
    float* upd   = csum  + BS;                         // [BS, D]
    float* gi    = upd   + (size_t)BS * Dq;            // [BS, 3D]
    float* gh    = gi    + (size_t)BS * 3 * Dq;        // [BS, 3D]
    float* t2    = gh    + (size_t)BS * 3 * Dq;        // [BS, 2D]
    // packed bf16 weights [N,K]
    __bf16* convp  = (__bf16*)(t2 + (size_t)BS * 2 * Dq);
    __bf16* w1p    = convp + 256 * Kpatch;             // pmlp_w1  [512,256]
    __bf16* w2p    = w1p   + 512 * Dq;                 // pmlp_w2  [256,512]
    __bf16* qwp    = w2p   + 256 * 2 * Dq;             // q_w      [256,256]
    __bf16* kwp    = qwp   + Dq * Dq;
    __bf16* vwp    = kwp   + Dq * Dq;
    __bf16* wihp   = vwp   + Dq * Dq;                  // gru_wih  [768,256]
    __bf16* whhp   = wihp  + 3 * Dq * Dq;
    __bf16* s1p    = whhp  + 3 * Dq * Dq;              // smlp_w1  [512,256]
    __bf16* s2p    = s1p   + 512 * Dq;                 // smlp_w2  [256,512]

    dim3 wave(32);
    auto PK = [&](const float* W, __bf16* Wp, int K, int N, int srcNK) {
        pack_w_kernel<<<(N * K + 255) / 256, 256, 0, stream>>>(W, Wp, K, N, srcNK);
    };

    // 0) pack weights to bf16 [N,K]
    PK(conv_w,  convp, Kpatch, 256, 1);
    PK(pmlp_w1, w1p,   Dq, 2 * Dq, 0);
    PK(pmlp_w2, w2p,   2 * Dq, Dq, 0);
    PK(q_w,     qwp,   Dq, Dq, 0);
    PK(k_w,     kwp,   Dq, Dq, 0);
    PK(v_w,     vwp,   Dq, Dq, 0);
    PK(gru_wih, wihp,  Dq, 3 * Dq, 1);
    PK(gru_whh, whhp,  Dq, 3 * Dq, 1);
    PK(smlp_w1, s1p,   Dq, 2 * Dq, 0);
    PK(smlp_w2, s2p,   2 * Dq, Dq, 0);

    // 1) patch embed + bias + pos  -> x0   (16x32 tiles)
    patch_embed_kernel<<<dim3(Dq / 32, BN / 16), wave, 0, stream>>>(
        state, convp, conv_b, pos, x0);

    // 2) pmlp: h = x0 @ w1 + b1 ; LN+GELU ; x = h @ w2 + b2 ; LN(ln_in)
    wmma_gemm_kernel<<<dim3(2 * Dq / 32, BN / 16), wave, 0, stream>>>(
        x0, w1p, pmlp_b1, nullptr, h, BN, 2 * Dq, Dq, 0);
    ln_kernel<<<BN, wave, 0, stream>>>(h, h, pmlp_lng, pmlp_lnb, 2 * Dq, 1);
    wmma_gemm_kernel<<<dim3(Dq / 32, BN / 16), wave, 0, stream>>>(
        h, w2p, pmlp_b2, nullptr, x0, BN, Dq, 2 * Dq, 0);
    ln_kernel<<<BN, wave, 0, stream>>>(x0, x0, ln_in_g, ln_in_b, Dq, 0);

    // 3) k, v projections
    wmma_gemm_kernel<<<dim3(Dq / 32, BN / 16), wave, 0, stream>>>(
        x0, kwp, nullptr, nullptr, kbuf, BN, Dq, Dq, 0);
    wmma_gemm_kernel<<<dim3(Dq / 32, BN / 16), wave, 0, stream>>>(
        x0, vwp, nullptr, nullptr, vbuf, BN, Dq, Dq, 0);

    // 4) slots = priors + mu
    init_slots_kernel<<<(BS * Dq + 255) / 256, 256, 0, stream>>>(priors, slot_mu, slots);

    // 5) slot-attention iterations (unrolled into the stream/graph)
    for (int it = 0; it < 3; ++it) {
        ln_kernel<<<BS, wave, 0, stream>>>(slots, snorm, ln_sl_g, ln_sl_b, Dq, 0);
        wmma_gemm_kernel<<<dim3(Dq / 32, BS / 16), wave, 0, stream>>>(
            snorm, qwp, nullptr, nullptr, qbuf, BS, Dq, Dq, 0);
        logits_kernel<<<dim3(Nq / 16, Bq), wave, 0, stream>>>(kbuf, qbuf, logit);
        softmax_kernel<<<(BN + 255) / 256, 256, 0, stream>>>(logit, attT);
        colsum_kernel<<<BS, 256, 0, stream>>>(attT, csum);
        upd_kernel<<<dim3(Dq / 16, Bq), wave, 0, stream>>>(attT, vbuf, csum, upd);
        // GRU gates (wih/whh packed [3D, D])
        wmma_gemm_kernel<<<dim3(3 * Dq / 32, BS / 16), wave, 0, stream>>>(
            upd, wihp, gru_bih, nullptr, gi, BS, 3 * Dq, Dq, 0);
        wmma_gemm_kernel<<<dim3(3 * Dq / 32, BS / 16), wave, 0, stream>>>(
            slots, whhp, gru_bhh, nullptr, gh, BS, 3 * Dq, Dq, 0);
        gru_kernel<<<(BS * Dq + 255) / 256, 256, 0, stream>>>(gi, gh, slots);
        // slot MLP residual: slots += gelu(LN(slots) @ w1 + b1) @ w2 + b2
        ln_kernel<<<BS, wave, 0, stream>>>(slots, snorm, ln_mlp_g, ln_mlp_b, Dq, 0);
        wmma_gemm_kernel<<<dim3(2 * Dq / 32, BS / 16), wave, 0, stream>>>(
            snorm, s1p, smlp_b1, nullptr, t2, BS, 2 * Dq, Dq, 1);
        wmma_gemm_kernel<<<dim3(Dq / 32, BS / 16), wave, 0, stream>>>(
            t2, s2p, smlp_b2, slots, slots, BS, Dq, 2 * Dq, 0);
    }

    // 6) emit slots (masks already in place in d_out from last softmax)
    copy_kernel<<<(BS * Dq + 255) / 256, 256, 0, stream>>>(slots, out_slots, BS * Dq);
}
